// InvariantPointAttention_63084479644125
// MI455X (gfx1250) — compile-verified, matrix operand order checked by reading, not run on hardware
//
#include <hip/hip_runtime.h>
#include <math.h>

typedef _Float16 half_t;
typedef half_t v16h __attribute__((ext_vector_type(16)));
typedef half_t v8h  __attribute__((ext_vector_type(8)));
typedef float  v8f  __attribute__((ext_vector_type(8)));

#define NRES 512
#define CSD  384
#define CZD  128
#define NH   12
#define NHP  16            /* heads padded to 16 for WMMA */
#define CH   16
#define NQPD 4
#define NPVD 8
#define NNIJ (NRES*NRES)
#define BC   48            /* padded columns: 16 (v) + 24 (wv pts) + 8 pad */
#define KCAT 2112          /* H*(CZ+C+4*NPV) = 12*176, divisible by 32 */

__device__ __forceinline__ v8f wmma_f16(v8f c, v16h a, v16h b) {
    return __builtin_amdgcn_wmma_f32_16x16x32_f16(false, a, false, b, (short)0, c, false, false);
}

// Fragment load: per-lane K elements are contiguous -> two 16B vector loads.
// base points at (row-or-col start + k_chunk_base); must be 16B aligned.
__device__ __forceinline__ v16h load_frag(const half_t* __restrict__ base, int hi) {
    v8h lo = *(const v8h*)(base + 8*hi);
    v8h hg = *(const v8h*)(base + 16 + 8*hi);
    return __builtin_shufflevector(lo, hg, 0,1,2,3,4,5,6,7,8,9,10,11,12,13,14,15);
}

// Same, but converting from contiguous fp32.
__device__ __forceinline__ v16h load_frag_f32(const float* __restrict__ base, int hi) {
    const float* p0 = base + 8*hi;
    const float* p1 = base + 16 + 8*hi;
    v16h r;
#pragma unroll
    for (int e = 0; e < 8; ++e) { r[e] = (half_t)p0[e]; r[e+8] = (half_t)p1[e]; }
    return r;
}

// ---------------- K0: z[i] (512j x 128c fp32) -> z16T[i] (128c x 512j f16) ----------------
__global__ void k_ztrans(const float* __restrict__ z, half_t* __restrict__ z16T) {
    __shared__ half_t tile[32][33];
    int i = blockIdx.x, j0 = blockIdx.y*32, c0 = blockIdx.z*32, t = threadIdx.x;
#pragma unroll
    for (int l = 0; l < 4; ++l) {
        int idx = t + l*256, jj = idx >> 5, cc = idx & 31;
        tile[jj][cc] = (half_t)z[((size_t)(i*NRES + j0 + jj))*CZD + c0 + cc];
    }
    __syncthreads();
#pragma unroll
    for (int l = 0; l < 4; ++l) {
        int idx = t + l*256, cc = idx >> 5, jj = idx & 31;
        z16T[((size_t)i*CZD + c0 + cc)*NRES + j0 + jj] = tile[jj][cc];
    }
}

// ---------------- K0b: WbT16[n][k] padded; K0c: WoT16[n][k] ----------------
__global__ void k_wbT(const float* __restrict__ Wb, half_t* __restrict__ WbT16) {
    for (int i = threadIdx.x; i < NHP*CZD; i += blockDim.x) {
        int n = i / CZD, k = i % CZD;
        WbT16[i] = (n < NH) ? (half_t)Wb[k*NH + n] : (half_t)0.f;
    }
}
__global__ void k_woT(const float* __restrict__ Wo, half_t* __restrict__ WoT16) {
    for (int i = blockIdx.x*blockDim.x + threadIdx.x; i < CSD*KCAT; i += gridDim.x*blockDim.x) {
        int n = i / KCAT, k = i % KCAT;
        WoT16[i] = (half_t)Wo[(size_t)k*CSD + n];
    }
}

// ---------------- K1: projections + frame warp ----------------
__global__ void k_proj(const float* __restrict__ s, const float* __restrict__ T,
                       const float* __restrict__ Wq, const float* __restrict__ bq,
                       const float* __restrict__ Wk, const float* __restrict__ bk,
                       const float* __restrict__ Wv, const float* __restrict__ bv,
                       const float* __restrict__ Wqp, const float* __restrict__ bqp,
                       const float* __restrict__ Wkp, const float* __restrict__ bkp,
                       const float* __restrict__ Wvp, const float* __restrict__ bvp,
                       float* __restrict__ qo, float* __restrict__ ko,
                       float* __restrict__ wq_pt, float* __restrict__ wk_pt,
                       float* __restrict__ qn, float* __restrict__ kn,
                       half_t* __restrict__ bcatT) {
    __shared__ float srow[CSD];
    __shared__ float pq[3*NH*NQPD], pk[3*NH*NQPD], pv[3*NH*NPVD];
    __shared__ float vrow[NH*CH];
    __shared__ float wqw[NH*NQPD*3], wkw[NH*NQPD*3];
    int n = blockIdx.x, t = threadIdx.x;
    for (int f = t; f < CSD; f += blockDim.x) srow[f] = s[n*CSD + f];
    __syncthreads();
    for (int o = t; o < 1152; o += blockDim.x) {
        const float* W; const float* bp; int col, outd;
        if (o < 192)      { W=Wq;  bp=bq;  col=o;     outd=192; }
        else if (o < 384) { W=Wk;  bp=bk;  col=o-192; outd=192; }
        else if (o < 576) { W=Wv;  bp=bv;  col=o-384; outd=192; }
        else if (o < 720) { W=Wqp; bp=bqp; col=o-576; outd=144; }
        else if (o < 864) { W=Wkp; bp=bkp; col=o-720; outd=144; }
        else              { W=Wvp; bp=bvp; col=o-864; outd=288; }
        float acc = bp[col];
        for (int f = 0; f < CSD; ++f) acc += srow[f]*W[f*outd + col];
        if (o < 192)      qo[n*192 + col] = acc;
        else if (o < 384) ko[n*192 + col] = acc;
        else if (o < 576) vrow[col] = acc;
        else if (o < 720) pq[col] = acc;
        else if (o < 864) pk[col] = acc;
        else              pv[col] = acc;
    }
    __syncthreads();
    float R[3][3], tt[3];
    for (int x = 0; x < 3; ++x) { for (int y = 0; y < 3; ++y) R[x][y] = T[n*16 + x*4 + y]; tt[x] = T[n*16 + x*4 + 3]; }
    for (int pt = t; pt < 192; pt += blockDim.x) {
        if (pt < 48) {                                   // q points: index = h*4+p
            float p0 = pq[pt], p1 = pq[48+pt], p2 = pq[96+pt];
            for (int x = 0; x < 3; ++x) {
                float w = R[x][0]*p0 + R[x][1]*p1 + R[x][2]*p2 + tt[x];
                wq_pt[(n*48 + pt)*3 + x] = w; wqw[pt*3 + x] = w;
            }
        } else if (pt < 96) {                            // k points
            int q = pt - 48;
            float p0 = pk[q], p1 = pk[48+q], p2 = pk[96+q];
            for (int x = 0; x < 3; ++x) {
                float w = R[x][0]*p0 + R[x][1]*p1 + R[x][2]*p2 + tt[x];
                wk_pt[(n*48 + q)*3 + x] = w; wkw[q*3 + x] = w;
            }
        } else {                                         // v points: 96 = H*NPV
            int q = pt - 96; int h = q/NPVD, p = q%NPVD;
            float p0 = pv[q], p1 = pv[96+q], p2 = pv[192+q];
            for (int x = 0; x < 3; ++x) {
                float w = R[x][0]*p0 + R[x][1]*p1 + R[x][2]*p2 + tt[x];
                bcatT[((size_t)h*BC + 16 + p*3 + x)*NRES + n] = (half_t)w;
            }
        }
    }
    __syncthreads();
    if (t < NH) {                                        // qn[h][n]
        float a = 0.f;
        for (int c = 0; c < 12; ++c) { float v = wqw[t*12 + c]; a += v*v; }
        qn[t*NRES + n] = a;
    } else if (t < 2*NH) {                               // kn[h][n]
        int h = t - NH; float a = 0.f;
        for (int c = 0; c < 12; ++c) { float v = wkw[h*12 + c]; a += v*v; }
        kn[h*NRES + n] = a;
    }
    for (int o = t; o < NH*BC; o += blockDim.x) {        // v-columns + zero pad (transposed)
        int h = o/BC, c = o%BC;
        if (c < 16)       bcatT[((size_t)h*BC + c)*NRES + n] = (half_t)vrow[h*CH + c];
        else if (c >= 40) bcatT[((size_t)h*BC + c)*NRES + n] = (half_t)0.f;
    }
}

// ---------------- K2: bias[h][ij] = (z @ Wb + bb) via WMMA ----------------
__global__ void k_bias(const float* __restrict__ z, const half_t* __restrict__ WbT16,
                       const float* __restrict__ bb, float* __restrict__ bias) {
    int wave = threadIdx.x >> 5, lane = threadIdx.x & 31;
    int l16 = lane & 15, hi = lane >> 4;
    int tile = blockIdx.x*(blockDim.x >> 5) + wave;      // 16 flat (i,j)-rows per tile
    size_t row = (size_t)tile*16 + l16;
    v8f acc = {};
#pragma unroll
    for (int kc = 0; kc < CZD/32; ++kc) {
        v16h a = load_frag_f32(z + row*CZD + kc*32, hi);
        v16h b = load_frag(WbT16 + l16*CZD + kc*32, hi);
        acc = wmma_f16(acc, a, b);
    }
    int h = l16;
    if (h < NH) {
        float bbv = bb[h];
#pragma unroll
        for (int r = 0; r < 8; ++r) {
            size_t rr = (size_t)tile*16 + hi*8 + r;
            bias[(size_t)h*NNIJ + rr] = acc[r] + bbv;
        }
    }
}

// ---------------- K3: logits + softmax -> att (f16, 16 padded heads) ----------------
__global__ void k_attn(const float* __restrict__ qo, const float* __restrict__ ko,
                       const float* __restrict__ wq_pt, const float* __restrict__ wk_pt,
                       const float* __restrict__ qn, const float* __restrict__ kn,
                       const float* __restrict__ bias, const float* __restrict__ hw,
                       half_t* __restrict__ att) {
    const float W_L = 0.57735026918962576f;              // sqrt(1/3)
    const float W_C = 0.23570226039551584f;              // sqrt(2/(9*NQP))
    int i = blockIdx.x, h = blockIdx.y, t = threadIdx.x;
    if (h >= NH) {                                       // zero pad heads (block-uniform)
        for (int j = t; j < NRES; j += blockDim.x)
            att[(size_t)h*NNIJ + (size_t)i*NRES + j] = (half_t)0.f;
        return;
    }
    __shared__ float qrow[CH], wqp[12], red[256], lg[NRES];
    __shared__ float qnv, gam;
    if (t < CH) qrow[t] = qo[i*192 + h*CH + t];
    if (t < 12) wqp[t] = wq_pt[(i*48 + h*4)*3 + t];
    if (t == 0) { qnv = qn[h*NRES + i]; gam = log1pf(expf(hw[h])); }
    __syncthreads();
    float lmax = -1e30f;
    for (int j = t; j < NRES; j += blockDim.x) {
        float qk = 0.f;
        for (int c = 0; c < CH; ++c) qk += qrow[c]*ko[j*192 + h*CH + c];
        qk *= 0.25f;                                     // / sqrt(C)
        float dp = 0.f;
        const float* wk = &wk_pt[(j*48 + h*4)*3];
        for (int c = 0; c < 12; ++c) dp += wqp[c]*wk[c];
        float d2 = qnv + kn[h*NRES + j] - 2.f*dp;
        float v = W_L*(qk + bias[(size_t)h*NNIJ + (size_t)i*NRES + j] - gam*W_C*0.5f*d2);
        lg[j] = v;
        lmax = fmaxf(lmax, v);
    }
    red[t] = lmax; __syncthreads();
    for (int s2 = 128; s2 > 0; s2 >>= 1) { if (t < s2) red[t] = fmaxf(red[t], red[t+s2]); __syncthreads(); }
    float mx = red[0]; __syncthreads();
    float lsum = 0.f;
    for (int j = t; j < NRES; j += blockDim.x) { float e = expf(lg[j] - mx); lg[j] = e; lsum += e; }
    red[t] = lsum; __syncthreads();
    for (int s2 = 128; s2 > 0; s2 >>= 1) { if (t < s2) red[t] += red[t+s2]; __syncthreads(); }
    float inv = 1.f/red[0];
    for (int j = t; j < NRES; j += blockDim.x)
        att[(size_t)h*NNIJ + (size_t)i*NRES + j] = (half_t)(lg[j]*inv);
}

// ---------------- K4: OV[h] = att[h] @ bcat[h]  (512x512 @ 512x48) ----------------
__global__ void k_av(const half_t* __restrict__ att, const half_t* __restrict__ bcatT,
                     float* __restrict__ OV) {
    int mt = blockIdx.x, nt = blockIdx.y, h = blockIdx.z;
    int lane = threadIdx.x & 31, l16 = lane & 15, hi = lane >> 4;
    int arow = mt*16 + l16, col = nt*16 + l16;
    const half_t* abase = att + (size_t)h*NNIJ + (size_t)arow*NRES;
    const half_t* bbase = bcatT + ((size_t)h*BC + col)*NRES;
    v8f acc = {};
#pragma unroll 4
    for (int kt = 0; kt < NRES/32; ++kt) {
        v16h a = load_frag(abase + kt*32, hi);
        v16h b = load_frag(bbase + kt*32, hi);
        acc = wmma_f16(acc, a, b);
    }
#pragma unroll
    for (int r = 0; r < 8; ++r) {
        int i = mt*16 + hi*8 + r;
        OV[(size_t)(h*NRES + i)*BC + col] = acc[r];
    }
}

// ---------------- K5: pairwise[i] = att[:,i,:] @ z[i]  (16x512 @ 512x128) ----------------
__global__ void k_pair(const half_t* __restrict__ att, const half_t* __restrict__ z16T,
                       float* __restrict__ pw) {
    int i = blockIdx.x, nt = blockIdx.y;
    int lane = threadIdx.x & 31, l16 = lane & 15, hi = lane >> 4;
    int col = nt*16 + l16;
    const half_t* abase = att + (size_t)l16*NNIJ + (size_t)i*NRES;  // padded heads are zero
    const half_t* bbase = z16T + ((size_t)i*CZD + col)*NRES;
    v8f acc = {};
#pragma unroll 4
    for (int kt = 0; kt < NRES/32; ++kt) {
        v16h a = load_frag(abase + kt*32, hi);
        v16h b = load_frag(bbase + kt*32, hi);
        acc = wmma_f16(acc, a, b);
    }
#pragma unroll
    for (int r = 0; r < 8; ++r) {
        int h = hi*8 + r;
        if (h < NH) pw[(size_t)(i*NH + h)*CZD + col] = acc[r];
    }
}

// ---------------- K6a: inverse warp + norms + concat -> cat16 ----------------
__global__ void k_cat(const float* __restrict__ OV, const float* __restrict__ pw,
                      const float* __restrict__ T, half_t* __restrict__ cat) {
    __shared__ float ol[3*96], nrm[96];
    int n = blockIdx.x, t = threadIdx.x;
    float R[3][3], tt[3];
    for (int x = 0; x < 3; ++x) { for (int y = 0; y < 3; ++y) R[x][y] = T[n*16 + x*4 + y]; tt[x] = T[n*16 + x*4 + 3]; }
    if (t < 96) {                                        // t = h*8 + p
        int h = t/NPVD, p = t%NPVD;
        float op[3];
        for (int x = 0; x < 3; ++x) op[x] = OV[(size_t)(h*NRES + n)*BC + 16 + p*3 + x] - tt[x];
        float s2 = 0.f;
        for (int x = 0; x < 3; ++x) {
            float v = R[0][x]*op[0] + R[1][x]*op[1] + R[2][x]*op[2];   // R^T
            ol[x*96 + t] = v; s2 += v*v;
        }
        nrm[t] = sqrtf(s2);
    }
    __syncthreads();
    for (int idx = t; idx < KCAT; idx += blockDim.x) {
        float v;
        if (idx < 192)      { int h = idx/CH, c = idx%CH; v = OV[(size_t)(h*NRES + n)*BC + c]; }
        else if (idx < 480) v = ol[idx - 192];           // x*96 + h*8 + p == reference 'bixhp'
        else if (idx < 576) v = nrm[idx - 480];
        else { int j = idx - 576; int h = j/CZD, c = j%CZD; v = pw[(size_t)(n*NH + h)*CZD + c]; }
        cat[(size_t)n*KCAT + idx] = (half_t)v;
    }
}

// ---------------- K6b: out = cat @ Wo + bo  (512x2112 @ 2112x384) ----------------
__global__ void k_out(const half_t* __restrict__ cat, const half_t* __restrict__ WoT16,
                      const float* __restrict__ bo, float* __restrict__ out) {
    int mt = blockIdx.x, nt = blockIdx.y;
    int lane = threadIdx.x & 31, l16 = lane & 15, hi = lane >> 4;
    int col = nt*16 + l16;
    const half_t* abase = cat + (size_t)(mt*16 + l16)*KCAT;
    const half_t* bbase = WoT16 + (size_t)col*KCAT;
    v8f acc = {};
#pragma unroll 2
    for (int kc = 0; kc < KCAT/32; ++kc) {
        v16h a = load_frag(abase + kc*32, hi);
        v16h b = load_frag(bbase + kc*32, hi);
        acc = wmma_f16(acc, a, b);
    }
#pragma unroll
    for (int r = 0; r < 8; ++r) {
        int row = mt*16 + hi*8 + r;
        out[(size_t)row*CSD + col] = acc[r] + bo[col];
    }
}

extern "C" void kernel_launch(void* const* d_in, const int* in_sizes, int n_in,
                              void* d_out, int out_size, void* d_ws, size_t ws_size,
                              hipStream_t stream) {
    (void)in_sizes; (void)n_in; (void)out_size; (void)ws_size;
    const float* s   = (const float*)d_in[0];
    const float* z   = (const float*)d_in[1];
    const float* T   = (const float*)d_in[2];
    const float* Wq  = (const float*)d_in[3];  const float* bq  = (const float*)d_in[4];
    const float* Wk  = (const float*)d_in[5];  const float* bk  = (const float*)d_in[6];
    const float* Wv  = (const float*)d_in[7];  const float* bv  = (const float*)d_in[8];
    const float* Wqp = (const float*)d_in[9];  const float* bqp = (const float*)d_in[10];
    const float* Wkp = (const float*)d_in[11]; const float* bkp = (const float*)d_in[12];
    const float* Wvp = (const float*)d_in[13]; const float* bvp = (const float*)d_in[14];
    const float* Wb  = (const float*)d_in[15]; const float* bb  = (const float*)d_in[16];
    const float* Wo  = (const float*)d_in[17]; const float* bo  = (const float*)d_in[18];
    const float* hw  = (const float*)d_in[19];
    float* out = (float*)d_out;

    size_t off = 0;
    auto carve = [&](size_t bytes) -> void* {
        void* p = (char*)d_ws + off;
        off += (bytes + 255) & ~(size_t)255;
        return p;
    };
    half_t* z16T  = (half_t*)carve((size_t)NNIJ*CZD*sizeof(half_t));       // 64 MB, transposed per i
    half_t* att   = (half_t*)carve((size_t)NHP*NNIJ*sizeof(half_t));       // 8 MB (16 padded heads)
    float*  bias  = (float*) carve((size_t)NH*NNIJ*sizeof(float));         // 12 MB
    half_t* WbT16 = (half_t*)carve((size_t)NHP*CZD*sizeof(half_t));
    half_t* WoT16 = (half_t*)carve((size_t)CSD*KCAT*sizeof(half_t));       // 1.6 MB
    float*  qo    = (float*) carve((size_t)NRES*NH*CH*sizeof(float));
    float*  ko    = (float*) carve((size_t)NRES*NH*CH*sizeof(float));
    float*  wqp   = (float*) carve((size_t)NRES*NH*NQPD*3*sizeof(float));
    float*  wkp   = (float*) carve((size_t)NRES*NH*NQPD*3*sizeof(float));
    float*  qn    = (float*) carve((size_t)NH*NRES*sizeof(float));
    float*  kn    = (float*) carve((size_t)NH*NRES*sizeof(float));
    half_t* bcatT = (half_t*)carve((size_t)NH*BC*NRES*sizeof(half_t));     // transposed [h][48][512]
    float*  OV    = (float*) carve((size_t)NH*NRES*BC*sizeof(float));
    float*  pw    = (float*) carve((size_t)NRES*NH*CZD*sizeof(float));
    half_t* cat   = (half_t*)carve((size_t)NRES*KCAT*sizeof(half_t));

    k_ztrans<<<dim3(NRES, NRES/32, CZD/32), 256, 0, stream>>>(z, z16T);
    k_wbT<<<1, 256, 0, stream>>>(Wb, WbT16);
    k_woT<<<1024, 256, 0, stream>>>(Wo, WoT16);
    k_proj<<<NRES, 256, 0, stream>>>(s, T, Wq, bq, Wk, bk, Wv, bv, Wqp, bqp, Wkp, bkp,
                                     Wvp, bvp, qo, ko, wqp, wkp, qn, kn, bcatT);
    k_bias<<<NNIJ/16/4, 128, 0, stream>>>(z, WbT16, bb, bias);
    k_attn<<<dim3(NRES, NHP), 256, 0, stream>>>(qo, ko, wqp, wkp, qn, kn, bias, hw, att);
    k_av<<<dim3(NRES/16, BC/16, NH), 32, 0, stream>>>(att, bcatT, OV);
    k_pair<<<dim3(NRES, CZD/16), 32, 0, stream>>>(att, z16T, pw);
    k_cat<<<NRES, 256, 0, stream>>>(OV, pw, T, cat);
    k_out<<<dim3(NRES/16, CSD/16), 32, 0, stream>>>(cat, WoT16, bo, out);
}
